// UNet_90185723281813
// MI455X (gfx1250) — matrix operand order repaired and checked by reading, not compiled, not run on hardware
//
#include <hip/hip_runtime.h>

typedef __attribute__((ext_vector_type(16))) __bf16        v16bf;
typedef __attribute__((ext_vector_type(8)))  float         v8f;
typedef __attribute__((ext_vector_type(8)))  unsigned int  v8u;
typedef int v4i_t __attribute__((vector_size(16)));

static constexpr int BATCH = 4;
static constexpr int N0 = 8192, N1 = 2048, N2 = 512, N3 = 128;

#if defined(__AMDGCN__) && __has_builtin(__builtin_amdgcn_global_load_async_to_lds_b128)
#define HAVE_ASYNC_LDS 1
#else
#define HAVE_ASYNC_LDS 0
#endif

__device__ __forceinline__ void wait_async_zero() {
#if defined(__AMDGCN__)
#if __has_builtin(__builtin_amdgcn_s_wait_asynccnt)
  __builtin_amdgcn_s_wait_asynccnt(0);
#else
  asm volatile("s_wait_asynccnt 0x0" ::: "memory");
#endif
#endif
}

// ---------------------------------------------------------------------------
// float -> bf16 (round-to-nearest-even)
__device__ __forceinline__ unsigned short f2bf(float f) {
  unsigned int u = __builtin_bit_cast(unsigned int, f);
  u += 0x7FFFu + ((u >> 16) & 1u);
  return (unsigned short)(u >> 16);
}

// ---------------------------------------------------------------------------
// Pack W (Kact x N, f32 row-major) into WMMA B-fragment order, bf16,
// K zero-padded to KP (multiple of 32).
// Linear dword index t = ((nt*KT + kt)*32 + lane)*8 + v
//   lane 0-15 : K = kt*32 + 2v, 2v+1       (N = nt*16 + lane)
//   lane 16-31: K = kt*32 + 16 + 2v, +1    (N = nt*16 + lane-16)
__global__ __launch_bounds__(256)
void pack_w_kernel(const float* __restrict__ W, unsigned int* __restrict__ out,
                   int Kact, int N, int KP) {
  const int KT = KP >> 5;
  const int total = KT * (N >> 4) * 256;
  const int t = blockIdx.x * 256 + threadIdx.x;
  if (t >= total) return;
  const int v    = t & 7;
  const int lane = (t >> 3) & 31;
  const int rest = t >> 8;
  const int kt   = rest % KT;
  const int nt   = rest / KT;
  const int n    = nt * 16 + (lane & 15);
  const int k0   = kt * 32 + ((lane < 16) ? 0 : 16) + v * 2;
  const float lo = (k0     < Kact) ? W[(size_t)k0       * N + n] : 0.f;
  const float hi = (k0 + 1 < Kact) ? W[(size_t)(k0 + 1) * N + n] : 0.f;
  out[t] = (unsigned)f2bf(lo) | ((unsigned)f2bf(hi) << 16);
}

// ---------------------------------------------------------------------------
// Brute-force KNN: query/ref points are both prefixes of pos0 (batch stride N0*3)
template <int KN>
__global__ __launch_bounds__(256)
void knn_kernel(const float* __restrict__ pos, int nq, int nr,
                int* __restrict__ idx_out, float* __restrict__ d2_out) {
  const float* P = pos + (size_t)blockIdx.y * (N0 * 3);
  const int q = blockIdx.x * 256 + threadIdx.x;
  float qx = 0.f, qy = 0.f, qz = 0.f;
  if (q < nq) { qx = P[q * 3 + 0]; qy = P[q * 3 + 1]; qz = P[q * 3 + 2]; }

  float bd[KN]; int bi[KN];
#pragma unroll
  for (int i = 0; i < KN; ++i) { bd[i] = 3.0e38f; bi[i] = 0; }

  __shared__ float tile[256 * 3];
  for (int base = 0; base < nr; base += 256) {
    const int cnt = (nr - base < 256) ? (nr - base) : 256;
    __syncthreads();
    for (int i = threadIdx.x; i < cnt * 3; i += 256) tile[i] = P[base * 3 + i];
    __syncthreads();
    if (q < nq) {
      for (int j = 0; j < cnt; ++j) {
        const float dx = tile[j * 3 + 0] - qx;
        const float dy = tile[j * 3 + 1] - qy;
        const float dz = tile[j * 3 + 2] - qz;
        const float d = dx * dx + dy * dy + dz * dz;
        if (d < bd[KN - 1]) {
          int p = KN - 1;
          while (p > 0 && bd[p - 1] > d) {
            bd[p] = bd[p - 1]; bi[p] = bi[p - 1]; --p;
          }
          bd[p] = d; bi[p] = base + j;
        }
      }
    }
  }
  if (q < nq) {
    int* I = idx_out + ((size_t)blockIdx.y * nq + q) * KN;
#pragma unroll
    for (int i = 0; i < KN; ++i) I[i] = bi[i];
    if (d2_out) {
      float* D = d2_out + ((size_t)blockIdx.y * nq + q) * KN;
#pragma unroll
      for (int i = 0; i < KN; ++i) D[i] = fmaxf(bd[i], 0.f);
    }
  }
}

// ---------------------------------------------------------------------------
// Build down-stage A matrix rows: row = (b*nc + m)*16 + k
// A[row] = [rel_pos(3) | feat(C)] bf16, zero padded to KP
__global__ __launch_bounds__(256)
void gather_down_kernel(const float* __restrict__ pos,
                        const float* __restrict__ feat, int nfeat, int C,
                        const int* __restrict__ idx, int nc,
                        unsigned short* __restrict__ A, int KP, int totalRows) {
  const int row = blockIdx.x * 256 + threadIdx.x;
  if (row >= totalRows) return;
  const int b = row / (nc * 16);
  const int m = (row >> 4) % nc;
  const float* P = pos + (size_t)b * (N0 * 3);
  const int j = idx[row];
  unsigned short* dst = A + (size_t)row * KP;
  dst[0] = f2bf(P[j * 3 + 0] - P[m * 3 + 0]);
  dst[1] = f2bf(P[j * 3 + 1] - P[m * 3 + 1]);
  dst[2] = f2bf(P[j * 3 + 2] - P[m * 3 + 2]);
  const float* F = feat + ((size_t)b * nfeat + j) * C;
  for (int c = 0; c < C; ++c) dst[3 + c] = f2bf(F[c]);
  for (int c = 3 + C; c < KP; ++c) dst[c] = 0;
}

// ---------------------------------------------------------------------------
// Up-stage: 3-NN inverse-distance interpolation + concat prev features.
// row = b*np + m ; A[row] = [interp(Cc) | prv(Cp) | prv2(Cp2)] bf16, pad to KP
__global__ __launch_bounds__(256)
void up_build_kernel(const float* __restrict__ xc, int nc, int Cc,
                     const int* __restrict__ idx, const float* __restrict__ d2,
                     const float* __restrict__ prv, int Cp,
                     const float* __restrict__ prv2, int Cp2,
                     int np, unsigned short* __restrict__ A, int KP,
                     int totalRows) {
  const int row = blockIdx.x * 256 + threadIdx.x;
  if (row >= totalRows) return;
  const int b = row / np;
  const int m = row % np;
  const int* I = idx + (size_t)row * 3;
  const float* D = d2 + (size_t)row * 3;
  float w0 = 1.f / (D[0] + 1e-8f);
  float w1 = 1.f / (D[1] + 1e-8f);
  float w2 = 1.f / (D[2] + 1e-8f);
  const float s = 1.f / (w0 + w1 + w2);
  w0 *= s; w1 *= s; w2 *= s;
  const float* X = xc + (size_t)b * nc * Cc;
  const float* p0 = X + (size_t)I[0] * Cc;
  const float* p1 = X + (size_t)I[1] * Cc;
  const float* p2 = X + (size_t)I[2] * Cc;
  unsigned short* dst = A + (size_t)row * KP;
  for (int c = 0; c < Cc; ++c)
    dst[c] = f2bf(w0 * p0[c] + w1 * p1[c] + w2 * p2[c]);
  const float* Pv = prv + ((size_t)b * np + m) * Cp;
  for (int c = 0; c < Cp; ++c) dst[Cc + c] = f2bf(Pv[c]);
  int o = Cc + Cp;
  if (prv2) {
    const float* P2 = prv2 + ((size_t)b * np + m) * Cp2;
    for (int c = 0; c < Cp2; ++c) dst[o + c] = f2bf(P2[c]);
    o += Cp2;
  }
  for (int c = o; c < KP; ++c) dst[c] = 0;
}

// ---------------------------------------------------------------------------
// WMMA GEMM: out(MxN) = A(MxKP bf16 row-major) * Wpacked(KPxN) + bias
// Block = 128 threads = 4 waves; block computes 32(M) x 128(N).
// Each wave owns a 2x2 grid of 16x16 tiles (2 M-tiles x 2 N-tiles) -> 4
// v_wmma_f32_16x16x32_bf16 per k-step, A frags reused across N, B across M.
// A tile (32 rows x 32 bf16) staged via one async b128 per thread (ASYNCcnt)
// or a vectorized uint4 copy when the async builtin is unavailable.
// MAXPOOL16: rows of an M-tile are the 16 neighbors of one center -> fused max.
template <int RELU, int OUT_BF16, int MAXPOOL16>
__global__ __launch_bounds__(128)
void wmma_gemm_kernel(const unsigned short* __restrict__ A,
                      const unsigned int* __restrict__ Wp,
                      const float* __restrict__ bias,
                      void* __restrict__ outp, int M, int N, int KP) {
  const int KT   = KP >> 5;
  const int mt0  = blockIdx.x * 2;             // first of two M tiles
  const int wave = threadIdx.x >> 5;
  const int lane = threadIdx.x & 31;
  const int nt0  = blockIdx.y * 8 + wave * 2;  // first of two N tiles

  __shared__ alignas(16) unsigned int As[512]; // 32 rows x 16 dwords

  v8f c[2][2];
#pragma unroll
  for (int ni = 0; ni < 2; ++ni) {
    const float bv = bias[(nt0 + ni) * 16 + (lane & 15)];
#pragma unroll
    for (int mi = 0; mi < 2; ++mi)
#pragma unroll
      for (int r = 0; r < 8; ++r) c[mi][ni][r] = bv;
  }

  const unsigned int* Ad = (const unsigned int*)A;
  const int rowDW = KP >> 1;                   // dwords per A row (mult of 4)
  const int sr = threadIdx.x >> 2;             // staging row 0..31
  const int sd = (threadIdx.x & 3) << 2;       // staging dword 0,4,8,12
  const unsigned int* gstage =
      Ad + (size_t)(mt0 * 16 + sr) * rowDW + sd;
  unsigned int* lstage = &As[sr * 16 + sd];

  for (int kt = 0; kt < KT; ++kt) {
    __syncthreads();
#if HAVE_ASYNC_LDS
    __builtin_amdgcn_global_load_async_to_lds_b128(
        (__attribute__((address_space(1))) v4i_t*)(gstage + (kt << 4)),
        (__attribute__((address_space(3))) v4i_t*)lstage, 0, 0);
    wait_async_zero();
#else
    *(uint4*)lstage = *(const uint4*)(gstage + (kt << 4));
#endif
    if (kt + 1 < KT)  // hint next A tile (global_prefetch_b8)
      __builtin_prefetch(gstage + ((kt + 1) << 4), 0, 1);
    __syncthreads();

    // A fragments: contiguous K order (k = 16*half + s), matching packed B
    const int kb = (lane < 16) ? 0 : 8;
    v16bf a[2];
#pragma unroll
    for (int mi = 0; mi < 2; ++mi) {
      const int arow = (mi * 16 + (lane & 15)) << 4;
      v8u au;
#pragma unroll
      for (int v = 0; v < 4; ++v) {
        au[v]     = As[arow + kb + v];
        au[v + 4] = As[arow + kb + 4 + v];
      }
      a[mi] = __builtin_bit_cast(v16bf, au);
    }
    // B fragments: pre-packed, 8 contiguous dwords per lane
    v16bf b[2];
#pragma unroll
    for (int ni = 0; ni < 2; ++ni) {
      const v8u bu =
          *(const v8u*)(Wp + (((size_t)(nt0 + ni) * KT + kt) * 32 + lane) * 8);
      b[ni] = __builtin_bit_cast(v16bf, bu);
    }
#pragma unroll
    for (int mi = 0; mi < 2; ++mi)
#pragma unroll
      for (int ni = 0; ni < 2; ++ni)
        c[mi][ni] = __builtin_amdgcn_wmma_f32_16x16x32_bf16(
            false, a[mi], false, b[ni], (short)0, c[mi][ni], false, false);
  }

  if (MAXPOOL16) {
#pragma unroll
    for (int mi = 0; mi < 2; ++mi)
#pragma unroll
      for (int ni = 0; ni < 2; ++ni) {
        float m8 = c[mi][ni][0];
#pragma unroll
        for (int r = 1; r < 8; ++r) m8 = fmaxf(m8, c[mi][ni][r]);
        if (RELU) m8 = fmaxf(m8, 0.f);
        m8 = fmaxf(m8, __shfl_xor(m8, 16, 32));  // M=0..7 with M=8..15
        if (lane < 16)
          ((float*)outp)[(size_t)(mt0 + mi) * N + (nt0 + ni) * 16 + lane] = m8;
      }
  } else {
#pragma unroll
    for (int mi = 0; mi < 2; ++mi) {
      const int mb = (mt0 + mi) * 16 + ((lane < 16) ? 0 : 8);
#pragma unroll
      for (int ni = 0; ni < 2; ++ni) {
        const int n = (nt0 + ni) * 16 + (lane & 15);
#pragma unroll
        for (int r = 0; r < 8; ++r) {
          float f = c[mi][ni][r];
          if (RELU) f = fmaxf(f, 0.f);
          if (OUT_BF16)
            ((unsigned short*)outp)[(size_t)(mb + r) * N + n] = f2bf(f);
          else
            ((float*)outp)[(size_t)(mb + r) * N + n] = f;
        }
      }
    }
  }
}

// ---------------------------------------------------------------------------
__global__ __launch_bounds__(256)
void cvt_f2bf_kernel(const float* __restrict__ src,
                     unsigned short* __restrict__ dst, int n) {
  const int i = blockIdx.x * 256 + threadIdx.x;
  if (i < n) dst[i] = f2bf(src[i]);
}

__global__ __launch_bounds__(256)
void copy_f32_kernel(const float* __restrict__ src, float* __restrict__ dst,
                     int n) {
  const int i = blockIdx.x * 256 + threadIdx.x;
  if (i < n) dst[i] = src[i];
}

// ---------------------------------------------------------------------------
extern "C" void kernel_launch(void* const* d_in, const int* in_sizes, int n_in,
                              void* d_out, int out_size, void* d_ws,
                              size_t ws_size, hipStream_t stream) {
  (void)in_sizes; (void)n_in; (void)out_size; (void)ws_size;

  const float* x    = (const float*)d_in[0];
  const float* pos  = (const float*)d_in[1];
  const float* d0W1 = (const float*)d_in[2];
  const float* d0b1 = (const float*)d_in[3];
  const float* d0W2 = (const float*)d_in[4];
  const float* d0b2 = (const float*)d_in[5];
  const float* d1W1 = (const float*)d_in[6];
  const float* d1b1 = (const float*)d_in[7];
  const float* d1W2 = (const float*)d_in[8];
  const float* d1b2 = (const float*)d_in[9];
  const float* d2W1 = (const float*)d_in[10];
  const float* d2b1 = (const float*)d_in[11];
  const float* d2W2 = (const float*)d_in[12];
  const float* d2b2 = (const float*)d_in[13];
  const float* u0W  = (const float*)d_in[14];
  const float* u0b  = (const float*)d_in[15];
  const float* u1W  = (const float*)d_in[16];
  const float* u1b  = (const float*)d_in[17];
  const float* u2W  = (const float*)d_in[18];
  const float* u2b  = (const float*)d_in[19];
  const float* fW1  = (const float*)d_in[20];
  const float* fb1  = (const float*)d_in[21];
  const float* fW2  = (const float*)d_in[22];
  const float* fb2  = (const float*)d_in[23];
  float* out = (float*)d_out;

  char* ws = (char*)d_ws;
  size_t off = 0;
  auto alloc = [&](size_t bytes) -> void* {
    off = (off + 255) & ~(size_t)255;
    void* p = ws + off;
    off += bytes;
    return p;
  };
  auto pack = [&](const float* W, int Kact, int N, int KP) -> unsigned int* {
    unsigned int* p = (unsigned int*)alloc((size_t)KP * N * 2);
    const int total = (KP / 32) * (N / 16) * 256;
    pack_w_kernel<<<(total + 255) / 256, 256, 0, stream>>>(W, p, Kact, N, KP);
    return p;
  };

  // ---- packed weights (persistent) ----
  unsigned int* wpD0W1 = pack(d0W1, 6,   128, 32);
  unsigned int* wpD0W2 = pack(d0W2, 128, 128, 128);
  unsigned int* wpD1W1 = pack(d1W1, 131, 256, 160);
  unsigned int* wpD1W2 = pack(d1W2, 256, 256, 256);
  unsigned int* wpD2W1 = pack(d2W1, 259, 512, 288);
  unsigned int* wpD2W2 = pack(d2W2, 512, 512, 512);
  unsigned int* wpU0   = pack(u0W,  768, 256, 768);
  unsigned int* wpU1   = pack(u1W,  384, 128, 384);
  unsigned int* wpU2   = pack(u2W,  134, 128, 160);
  unsigned int* wpF1   = pack(fW1,  128, 128, 128);
  unsigned int* wpF2   = pack(fW2,  128, 128, 128);

  // ---- persistent features ----
  float* x1  = (float*)alloc((size_t)BATCH * N1 * 128 * 4);
  float* x2  = (float*)alloc((size_t)BATCH * N2 * 256 * 4);
  float* x3  = (float*)alloc((size_t)BATCH * N3 * 512 * 4);
  float* up0 = (float*)alloc((size_t)BATCH * N2 * 256 * 4);
  float* up1 = (float*)alloc((size_t)BATCH * N1 * 128 * 4);
  float* up2 = (float*)alloc((size_t)BATCH * N0 * 128 * 4);

  int*   iD0 = (int*)alloc((size_t)BATCH * N1 * 16 * 4);
  int*   iD1 = (int*)alloc((size_t)BATCH * N2 * 16 * 4);
  int*   iD2 = (int*)alloc((size_t)BATCH * N3 * 16 * 4);
  int*   iU0 = (int*)alloc((size_t)BATCH * N2 * 3 * 4);
  float* dU0 = (float*)alloc((size_t)BATCH * N2 * 3 * 4);
  int*   iU1 = (int*)alloc((size_t)BATCH * N1 * 3 * 4);
  float* dU1 = (float*)alloc((size_t)BATCH * N1 * 3 * 4);
  int*   iU2 = (int*)alloc((size_t)BATCH * N0 * 3 * 4);
  float* dU2 = (float*)alloc((size_t)BATCH * N0 * 3 * 4);

  // ---- transient (reused across stages) ----
  unsigned short* Ab = (unsigned short*)alloc((size_t)5242880 * 2);   // max A
  unsigned short* Hb = (unsigned short*)alloc((size_t)16777216 * 2);  // max H

  auto knn16 = [&](int nq, int nr, int* idx) {
    knn_kernel<16><<<dim3((nq + 255) / 256, BATCH), 256, 0, stream>>>(
        pos, nq, nr, idx, nullptr);
  };
  auto knn3 = [&](int nq, int nr, int* idx, float* d2) {
    knn_kernel<3><<<dim3((nq + 255) / 256, BATCH), 256, 0, stream>>>(
        pos, nq, nr, idx, d2);
  };

  // ======================= down stage 0 =======================
  {
    const int M = BATCH * N1 * 16;
    knn16(N1, N0, iD0);
    gather_down_kernel<<<(M + 255) / 256, 256, 0, stream>>>(
        pos, x, N0, 3, iD0, N1, Ab, 32, M);
    wmma_gemm_kernel<1, 1, 0><<<dim3(M / 32, 128 / 128), 128, 0, stream>>>(
        Ab, wpD0W1, d0b1, Hb, M, 128, 32);
    wmma_gemm_kernel<0, 0, 1><<<dim3(M / 32, 128 / 128), 128, 0, stream>>>(
        Hb, wpD0W2, d0b2, x1, M, 128, 128);
  }
  // ======================= down stage 1 =======================
  {
    const int M = BATCH * N2 * 16;
    knn16(N2, N1, iD1);
    gather_down_kernel<<<(M + 255) / 256, 256, 0, stream>>>(
        pos, x1, N1, 128, iD1, N2, Ab, 160, M);
    wmma_gemm_kernel<1, 1, 0><<<dim3(M / 32, 256 / 128), 128, 0, stream>>>(
        Ab, wpD1W1, d1b1, Hb, M, 256, 160);
    wmma_gemm_kernel<0, 0, 1><<<dim3(M / 32, 256 / 128), 128, 0, stream>>>(
        Hb, wpD1W2, d1b2, x2, M, 256, 256);
  }
  // ======================= down stage 2 =======================
  {
    const int M = BATCH * N3 * 16;
    knn16(N3, N2, iD2);
    gather_down_kernel<<<(M + 255) / 256, 256, 0, stream>>>(
        pos, x2, N2, 256, iD2, N3, Ab, 288, M);
    wmma_gemm_kernel<1, 1, 0><<<dim3(M / 32, 512 / 128), 128, 0, stream>>>(
        Ab, wpD2W1, d2b1, Hb, M, 512, 288);
    wmma_gemm_kernel<0, 0, 1><<<dim3(M / 32, 512 / 128), 128, 0, stream>>>(
        Hb, wpD2W2, d2b2, x3, M, 512, 512);
  }
  // ======================= up stage 0 =========================
  {
    const int M = BATCH * N2;
    knn3(N2, N3, iU0, dU0);
    up_build_kernel<<<(M + 255) / 256, 256, 0, stream>>>(
        x3, N3, 512, iU0, dU0, x2, 256, nullptr, 0, N2, Ab, 768, M);
    wmma_gemm_kernel<1, 0, 0><<<dim3(M / 32, 256 / 128), 128, 0, stream>>>(
        Ab, wpU0, u0b, up0, M, 256, 768);
  }
  // ======================= up stage 1 =========================
  {
    const int M = BATCH * N1;
    knn3(N1, N2, iU1, dU1);
    up_build_kernel<<<(M + 255) / 256, 256, 0, stream>>>(
        up0, N2, 256, iU1, dU1, x1, 128, nullptr, 0, N1, Ab, 384, M);
    wmma_gemm_kernel<1, 0, 0><<<dim3(M / 32, 128 / 128), 128, 0, stream>>>(
        Ab, wpU1, u1b, up1, M, 128, 384);
  }
  // ======================= up stage 2 =========================
  {
    const int M = BATCH * N0;
    knn3(N0, N1, iU2, dU2);
    up_build_kernel<<<(M + 255) / 256, 256, 0, stream>>>(
        up1, N1, 128, iU2, dU2, x, 3, pos, 3, N0, Ab, 160, M);
    wmma_gemm_kernel<1, 0, 0><<<dim3(M / 32, 128 / 128), 128, 0, stream>>>(
        Ab, wpU2, u2b, up2, M, 128, 160);
  }
  // ======================= head ===============================
  {
    const int M = BATCH * N0;
    const int n = M * 128;
    cvt_f2bf_kernel<<<(n + 255) / 256, 256, 0, stream>>>(up2, Ab, n);
    wmma_gemm_kernel<1, 1, 0><<<dim3(M / 32, 128 / 128), 128, 0, stream>>>(
        Ab, wpF1, fb1, Hb, M, 128, 128);
    wmma_gemm_kernel<0, 0, 0><<<dim3(M / 32, 128 / 128), 128, 0, stream>>>(
        Hb, wpF2, fb2, out, M, 128, 128);
  }
  // tuple output: (out, pos0) concatenated flat
  {
    const int n = BATCH * N0 * 3;
    copy_f32_kernel<<<(n + 255) / 256, 256, 0, stream>>>(
        pos, out + (size_t)BATCH * N0 * 128, n);
  }
}
